// VectorQuantizer_60610578481190
// MI455X (gfx1250) — matrix operand order repaired and checked by reading, not compiled, not run on hardware
//
#include <hip/hip_runtime.h>
#include <hip/hip_bf16.h>
#include <stdint.h>

// ---------------------------------------------------------------------------
// VQ-VAE vector quantizer for MI455X (gfx1250, wave32, WMMA).
//
// d = ||x||^2 + ||e||^2 - 2 x.e ; argmin over 8192 codes; z_q gather; loss.
// GEMM runs as bf16 hi/lo split (x ~= hi + lo) on v_wmma_f32_16x16x32_bf16
// with f32 accumulation: hi*hi + lo*hi + hi*lo  => ~f32-accurate dot products.
// Codebook tiles stream global->LDS via async loads (ASYNCcnt), double
// buffered so L2 latency overlaps the WMMA burst.
// ---------------------------------------------------------------------------

typedef __bf16 bf16_t;
typedef __attribute__((ext_vector_type(8)))  __bf16 v8bf;
typedef __attribute__((ext_vector_type(16))) __bf16 v16bf;
typedef __attribute__((ext_vector_type(8)))  float  v8f;

union V16 { v16bf v; v8bf h[2]; };

#define N_ROWS   16384     // 16*32*32
#define N_CODES  8192
#define C_DIM    256
#define N_ELEMS  (N_ROWS * C_DIM)   // 4194304
#define VTILE    16        // codes per LDS tile (double buffered)
#define NTILES   (N_CODES / VTILE)  // 512
#define BSTRIDE  264       // padded LDS row stride (bf16 elems): 528B = 33*16B
#define ROWS_PER_BLOCK 128 // 8 waves * 16 rows
#define LOSS_BLOCKS 1024

// ---------------------------------------------------------------------------
// 1) codebook: bf16 hi/lo split + squared norms
// ---------------------------------------------------------------------------
__global__ __launch_bounds__(256) void prep_codebook(
    const float* __restrict__ cb, bf16_t* __restrict__ cb_hi,
    bf16_t* __restrict__ cb_lo, float* __restrict__ cbnorm) {
  __shared__ float red[256];
  const int v = blockIdx.x;
  const int k = threadIdx.x;
  const float e = cb[(size_t)v * C_DIM + k];
  const bf16_t h = (bf16_t)e;
  cb_hi[(size_t)v * C_DIM + k] = h;
  cb_lo[(size_t)v * C_DIM + k] = (bf16_t)(e - (float)h);
  red[k] = e * e;
  __syncthreads();
  for (int s = 128; s > 0; s >>= 1) {
    if (k < s) red[k] += red[k + s];
    __syncthreads();
  }
  if (k == 0) cbnorm[v] = red[0];
}

// ---------------------------------------------------------------------------
// 2) inputs: bf16 hi/lo split
// ---------------------------------------------------------------------------
__global__ __launch_bounds__(256) void prep_inputs(
    const float* __restrict__ x, bf16_t* __restrict__ xh,
    bf16_t* __restrict__ xl, int n) {
  const int i = blockIdx.x * blockDim.x + threadIdx.x;
  if (i < n) {
    const float e = x[i];
    const bf16_t h = (bf16_t)e;
    xh[i] = h;
    xl[i] = (bf16_t)(e - (float)h);
  }
}

// ---------------------------------------------------------------------------
// Async global -> LDS tile fetch (GVS mode: saddr base + 32-bit voffset).
// Per call each thread issues 4 global_load_async_to_lds_b128 (ASYNCcnt +4).
// LDS byte address = low 32 bits of the generic pointer (ISA 10.2: LDS
// aperture keeps the allocation offset in addr[31:0]).
// ---------------------------------------------------------------------------
__device__ __forceinline__ void async_tile_load(
    const bf16_t* __restrict__ gb_hi, const bf16_t* __restrict__ gb_lo,
    bf16_t* lds_hi, bf16_t* lds_lo, int vbase, int tid) {
#pragma unroll
  for (int it = 0; it < 2; ++it) {
    const int c = tid + it * 256;      // 512 chunks of 8 bf16 per precision
    const int code = c >> 5;           // 0..15
    const int k = (c & 31) * 8;        // 0..248
    const uint32_t lds_hi_off =
        (uint32_t)(uintptr_t)(lds_hi + code * BSTRIDE + k);
    const uint32_t lds_lo_off =
        (uint32_t)(uintptr_t)(lds_lo + code * BSTRIDE + k);
    const uint32_t goff =
        (uint32_t)((uint32_t)(vbase + code) * C_DIM + (uint32_t)k) *
        (uint32_t)sizeof(bf16_t);
    asm volatile("global_load_async_to_lds_b128 %0, %1, %2"
                 :
                 : "v"(lds_hi_off), "v"(goff),
                   "s"((uint64_t)(uintptr_t)gb_hi)
                 : "memory");
    asm volatile("global_load_async_to_lds_b128 %0, %1, %2"
                 :
                 : "v"(lds_lo_off), "v"(goff),
                   "s"((uint64_t)(uintptr_t)gb_lo)
                 : "memory");
  }
}

// ---------------------------------------------------------------------------
// 3) fused GEMM + argmin.
//    Block: 256 threads = 8 waves; each wave owns 16 rows (A in registers),
//    all waves share a 16-code B tile, async-loaded + double buffered in LDS.
// ---------------------------------------------------------------------------
__global__ __launch_bounds__(256) void vq_main(
    const bf16_t* __restrict__ in_hi, const bf16_t* __restrict__ in_lo,
    const bf16_t* __restrict__ cb_hi, const bf16_t* __restrict__ cb_lo,
    const float* __restrict__ cbnorm, int* __restrict__ best_idx_out,
    float* __restrict__ idx_out_f) {
  // [buffer][hi/lo][code*BSTRIDE + k] : 2*2*16*264*2B = 33792 B
  __shared__ __attribute__((aligned(16))) bf16_t sB[2][2][VTILE * BSTRIDE];

  const int tid  = threadIdx.x;
  const int lane = tid & 31;
  const int wave = tid >> 5;
  const int cls  = lane & 15;   // column class / A row-in-tile (ISA layout)
  const int half = lane >> 4;   // K-half select (ISA layout)
  const int rowbase = blockIdx.x * ROWS_PER_BLOCK + wave * 16;

  // --- A fragments (16 rows x 256 K, hi+lo) resident in registers: 128 VGPRs.
  // 16-bit A layout: lane(m=cls, half): VGPR0..3 hold K = 8*half + 0..7,
  // VGPR4..7 hold K = 16 + 8*half + 0..7 (per 32-K step).
  V16 a_hi[8], a_lo[8];
  {
    const bf16_t* ah = in_hi + (size_t)(rowbase + cls) * C_DIM;
    const bf16_t* al = in_lo + (size_t)(rowbase + cls) * C_DIM;
#pragma unroll
    for (int ks = 0; ks < 8; ++ks) {
      const int kb = ks * 32 + half * 8;
      a_hi[ks].h[0] = *(const v8bf*)(ah + kb);
      a_hi[ks].h[1] = *(const v8bf*)(ah + kb + 16);
      a_lo[ks].h[0] = *(const v8bf*)(al + kb);
      a_lo[ks].h[1] = *(const v8bf*)(al + kb + 16);
    }
  }

  float best[8];
  int   bidx[8];
#pragma unroll
  for (int r = 0; r < 8; ++r) { best[r] = 3.402823466e38f; bidx[r] = 0; }

  // Prologue: async-fetch tile 0 into buffer 0.
  async_tile_load(cb_hi, cb_lo, &sB[0][0][0], &sB[0][1][0], 0, tid);

  for (int t = 0; t < NTILES; ++t) {
    const int buf = t & 1;
    if (t + 1 < NTILES) {
      // Issue next tile into the other buffer (safe: end-of-iter barrier of
      // t-1 guarantees everyone finished computing from it), then wait for
      // this tile's 4 per-thread loads (async loads complete in order).
      async_tile_load(cb_hi, cb_lo, &sB[buf ^ 1][0][0], &sB[buf ^ 1][1][0],
                      (t + 1) * VTILE, tid);
      asm volatile("s_wait_asynccnt 0x4" ::: "memory");
    } else {
      asm volatile("s_wait_asynccnt 0x0" ::: "memory");
    }
    __syncthreads();  // all threads' tile-t data visible in LDS

    // B layout per 32-K step: lane(n=cls, half): VGPR0..3 hold K = 8*half +
    // 0..7, VGPR4..7 hold K = 16 + 8*half + 0..7 (matches A).
    const bf16_t* bh = &sB[buf][0][0] + cls * BSTRIDE + half * 8;
    const bf16_t* bl = &sB[buf][1][0] + cls * BSTRIDE + half * 8;
    v8f acc0 = {0.f, 0.f, 0.f, 0.f, 0.f, 0.f, 0.f, 0.f};
    v8f acc1 = {0.f, 0.f, 0.f, 0.f, 0.f, 0.f, 0.f, 0.f};
#pragma unroll
    for (int ks = 0; ks < 8; ks += 2) {
      V16 Bh0, Bl0, Bh1, Bl1;
      Bh0.h[0] = *(const v8bf*)(bh + ks * 32);
      Bh0.h[1] = *(const v8bf*)(bh + ks * 32 + 16);
      Bl0.h[0] = *(const v8bf*)(bl + ks * 32);
      Bl0.h[1] = *(const v8bf*)(bl + ks * 32 + 16);
      Bh1.h[0] = *(const v8bf*)(bh + (ks + 1) * 32);
      Bh1.h[1] = *(const v8bf*)(bh + (ks + 1) * 32 + 16);
      Bl1.h[0] = *(const v8bf*)(bl + (ks + 1) * 32);
      Bl1.h[1] = *(const v8bf*)(bl + (ks + 1) * 32 + 16);
      // Two independent accumulation chains for WMMA ILP.
      acc0 = __builtin_amdgcn_wmma_f32_16x16x32_bf16(
          false, a_hi[ks].v, false, Bh0.v, (short)0, acc0, false, false);
      acc1 = __builtin_amdgcn_wmma_f32_16x16x32_bf16(
          false, a_hi[ks + 1].v, false, Bh1.v, (short)0, acc1, false, false);
      acc0 = __builtin_amdgcn_wmma_f32_16x16x32_bf16(
          false, a_lo[ks].v, false, Bh0.v, (short)0, acc0, false, false);
      acc1 = __builtin_amdgcn_wmma_f32_16x16x32_bf16(
          false, a_lo[ks + 1].v, false, Bh1.v, (short)0, acc1, false, false);
      acc0 = __builtin_amdgcn_wmma_f32_16x16x32_bf16(
          false, a_hi[ks].v, false, Bl0.v, (short)0, acc0, false, false);
      acc1 = __builtin_amdgcn_wmma_f32_16x16x32_bf16(
          false, a_hi[ks + 1].v, false, Bl1.v, (short)0, acc1, false, false);
    }
    const int n_global = t * VTILE + cls;
    const float nrm = cbnorm[n_global];
#pragma unroll
    for (int r = 0; r < 8; ++r) {
      const float s = nrm - 2.0f * (acc0[r] + acc1[r]);  // ||x||^2 const/row
      if (s < best[r]) { best[r] = s; bidx[r] = n_global; }
    }
    __syncthreads();  // compute done; buffer may be overwritten next iter
  }

  // Cross-lane argmin over the 16 column classes (xor reduce within 16-group).
#pragma unroll
  for (int r = 0; r < 8; ++r) {
#pragma unroll
    for (int mask = 1; mask < 16; mask <<= 1) {
      const float ob = __shfl_xor(best[r], mask, 32);
      const int   oi = __shfl_xor(bidx[r], mask, 32);
      if (ob < best[r] || (ob == best[r] && oi < bidx[r])) {
        best[r] = ob; bidx[r] = oi;
      }
    }
  }
  if (cls == 0) {
#pragma unroll
    for (int r = 0; r < 8; ++r) {
      const int row = rowbase + r + 8 * half;  // D layout: VGPR r, half
      best_idx_out[row] = bidx[r];
      idx_out_f[row] = (float)bidx[r];
    }
  }
}

// ---------------------------------------------------------------------------
// 4) gather z_q (exact f32 codebook rows) + per-block loss partials
// ---------------------------------------------------------------------------
__global__ __launch_bounds__(256) void gather_loss(
    const float* __restrict__ x, const float* __restrict__ cb,
    const int* __restrict__ best_idx, float* __restrict__ zq_out,
    float* __restrict__ partials, int n) {
  __shared__ float red[256];
  float acc = 0.f;
  for (int i = blockIdx.x * blockDim.x + threadIdx.x; i < n;
       i += gridDim.x * blockDim.x) {
    const int row = i >> 8;
    const int k = i & 255;
    const int idx = best_idx[row];
    const float zq = cb[(size_t)idx * C_DIM + k];
    zq_out[i] = zq;  // z_q_st == z_q numerically (straight-through)
    const float d = zq - x[i];
    acc += d * d;
  }
  red[threadIdx.x] = acc;
  __syncthreads();
  for (int s = 128; s > 0; s >>= 1) {
    if (threadIdx.x < s) red[threadIdx.x] += red[threadIdx.x + s];
    __syncthreads();
  }
  if (threadIdx.x == 0) partials[blockIdx.x] = red[0];
}

// 5) deterministic final reduction: loss = 1.25 * mean((zq - x)^2)
__global__ __launch_bounds__(256) void finalize_loss(
    const float* __restrict__ partials, int np, float* __restrict__ loss_out) {
  __shared__ float red[256];
  float a = 0.f;
  for (int i = threadIdx.x; i < np; i += 256) a += partials[i];
  red[threadIdx.x] = a;
  __syncthreads();
  for (int s = 128; s > 0; s >>= 1) {
    if (threadIdx.x < s) red[threadIdx.x] += red[threadIdx.x + s];
    __syncthreads();
  }
  if (threadIdx.x == 0) *loss_out = red[0] * (1.25f / (float)N_ELEMS);
}

// ---------------------------------------------------------------------------
extern "C" void kernel_launch(void* const* d_in, const int* in_sizes, int n_in,
                              void* d_out, int out_size, void* d_ws,
                              size_t ws_size, hipStream_t stream) {
  const float* x  = (const float*)d_in[0];   // [16,32,32,256]
  const float* cb = (const float*)d_in[1];   // [8192,256]

  float* out   = (float*)d_out;
  float* zq    = out;               // [4194304] z_q_st
  float* loss  = out + N_ELEMS;     // [1]
  float* idxf  = out + N_ELEMS + 1; // [16384] indices (as float)

  char* ws = (char*)d_ws;
  bf16_t* cb_hi = (bf16_t*)ws;  ws += (size_t)N_CODES * C_DIM * sizeof(bf16_t);
  bf16_t* cb_lo = (bf16_t*)ws;  ws += (size_t)N_CODES * C_DIM * sizeof(bf16_t);
  bf16_t* in_hi = (bf16_t*)ws;  ws += (size_t)N_ROWS * C_DIM * sizeof(bf16_t);
  bf16_t* in_lo = (bf16_t*)ws;  ws += (size_t)N_ROWS * C_DIM * sizeof(bf16_t);
  float*  cbnorm = (float*)ws;  ws += (size_t)N_CODES * sizeof(float);
  int*    bestIdx = (int*)ws;   ws += (size_t)N_ROWS * sizeof(int);
  float*  partials = (float*)ws;

  prep_codebook<<<N_CODES, 256, 0, stream>>>(cb, cb_hi, cb_lo, cbnorm);
  prep_inputs<<<(N_ELEMS + 255) / 256, 256, 0, stream>>>(x, in_hi, in_lo,
                                                         N_ELEMS);
  vq_main<<<N_ROWS / ROWS_PER_BLOCK, 256, 0, stream>>>(
      in_hi, in_lo, cb_hi, cb_lo, cbnorm, bestIdx, idxf);
  gather_loss<<<LOSS_BLOCKS, 256, 0, stream>>>(x, cb, bestIdx, zq, partials,
                                               N_ELEMS);
  finalize_loss<<<1, 256, 0, stream>>>(partials, LOSS_BLOCKS, loss);
}
